// MHCA_79671643340817
// MI455X (gfx1250) — matrix operand order repaired and checked by reading, not compiled, not run on hardware
//
#include <hip/hip_runtime.h>

// ---------------------------------------------------------------------------
// MHCA for MI455X (gfx1250): bf16 WMMA pipeline
//   k0: convert weights fp32 -> bf16
//   k1: p = Wqkv * X + b        (GEMM M=512 K=512 N=65536, bf16 WMMA,
//                                double-buffered LDS, reg prefetch)
//   k2: per (b,head): att = p p^T /64, softmax, out = att p   (WMMA)
//   k3: y = Wproj * out + b     (GEMM, fp32 output)
// ---------------------------------------------------------------------------

typedef __attribute__((ext_vector_type(16))) __bf16 v16bf;
typedef __attribute__((ext_vector_type(8)))  float  v8f;

union Frag {
    uint4 q[2];
    v16bf v;
};
union U4S8 {
    uint4 q;
    unsigned short s[8];
};
union U2S4 {
    uint2 u;
    unsigned short s[4];
};

__device__ __forceinline__ unsigned short f2bf(float f) {
    unsigned u = __builtin_bit_cast(unsigned, f);
    unsigned r = u + 0x7FFFu + ((u >> 16) & 1u);   // round-to-nearest-even
    return (unsigned short)(r >> 16);
}

__device__ __forceinline__ v8f wmma_bf16(const Frag& a, const Frag& b, v8f c) {
    return __builtin_amdgcn_wmma_f32_16x16x32_bf16(
        /*neg_a=*/false, a.v, /*neg_b=*/false, b.v,
        /*c_mod=*/(short)0, c, /*reuse_a=*/false, /*reuse_b=*/false);
}

// ---------------------------------------------------------------------------
// fp32 -> bf16 streaming convert
// ---------------------------------------------------------------------------
__global__ __launch_bounds__(256) void cvt_bf16_kernel(
    const float* __restrict__ a, unsigned short* __restrict__ o, int n)
{
    int i = blockIdx.x * blockDim.x + threadIdx.x;
    if (i < n) o[i] = f2bf(a[i]);
}

// ---------------------------------------------------------------------------
// GEMM: Out[b, m, n] = sum_k Wbf[m,k] * B[b, k, n] + bias[m]
//   M=512, K=512, columns N = b*4096 + n  (N total = 65536)
//   Tile: BM=64, BN=128, BK=32. 256 threads = 8 waves, wave -> 32x32 output.
//   Double-buffered LDS, register prefetch, 1 barrier / K-step.
// ---------------------------------------------------------------------------
template<bool B_BF16, bool OUT_BF16>
__global__ __launch_bounds__(256) void gemm_wmma_kernel(
    const unsigned short* __restrict__ Wbf,  // 512x512 bf16 row-major
    const void* __restrict__ Bsrc,           // (16,512,4096) fp32 or bf16
    const float* __restrict__ bias,          // 512 fp32
    void* __restrict__ Out)                  // (16,512,4096) bf16 or fp32
{
    const int tid  = threadIdx.x;
    const int lane = tid & 31;
    const int wave = tid >> 5;
    const int lr   = lane & 15;
    const int hi   = lane >> 4;                 // 0/1

    const int mblock = blockIdx.y * 64;
    const int nblk   = blockIdx.x * 128;        // global column block
    const int bb     = nblk >> 12;              // batch
    const int nloc   = nblk & 4095;             // column within batch

    const int mw = (wave >> 2) * 32;            // wave M offset in tile
    const int nw = (wave & 3) * 32;             // wave N offset in tile

    __shared__ __align__(16) unsigned short Asm[2][64][40];    // [m][k]
    __shared__ __align__(16) unsigned short Bsm[2][128][40];   // [n][k] (transposed)

    const size_t bbase = ((size_t)bb * 512) * 4096 + (size_t)nloc;

    // staging assignments
    const int ar = (tid * 8) >> 5;              // A row (0..63)
    const int ac = (tid * 8) & 31;              // A col {0,8,16,24}
    const unsigned short* aSrc = Wbf + (size_t)(mblock + ar) * 512 + ac;
    const int bkp = tid >> 5;                   // B k-pair base (0..7), +8 half 1
    const int bn0 = (tid & 31) * 4;             // B n offset (0..124)

    uint4  aReg;
    float4 bRegF[4];
    uint2  bRegH[4];

    v8f acc[2][2] = {};

    auto g2r = [&](int k0) {
        aReg = *(const uint4*)(aSrc + k0);
#pragma unroll
        for (int h = 0; h < 2; ++h) {
            int k = k0 + (bkp + h * 8) * 2;
            size_t g0 = bbase + (size_t)k * 4096 + bn0;
            if (B_BF16) {
                const unsigned short* bs = (const unsigned short*)Bsrc;
                bRegH[h * 2 + 0] = *(const uint2*)(bs + g0);          // row k
                bRegH[h * 2 + 1] = *(const uint2*)(bs + g0 + 4096);   // row k+1
            } else {
                const float* bs = (const float*)Bsrc;
                bRegF[h * 2 + 0] = *(const float4*)(bs + g0);
                bRegF[h * 2 + 1] = *(const float4*)(bs + g0 + 4096);
            }
        }
    };

    auto r2s = [&](int buf) {
        *(uint4*)&Asm[buf][ar][ac] = aReg;
#pragma unroll
        for (int h = 0; h < 2; ++h) {
            int kp = bkp + h * 8;
            if (B_BF16) {
                U2S4 r0, r1;
                r0.u = bRegH[h * 2 + 0];
                r1.u = bRegH[h * 2 + 1];
#pragma unroll
                for (int i = 0; i < 4; ++i) {
                    unsigned v = (unsigned)r0.s[i] | ((unsigned)r1.s[i] << 16);
                    *(unsigned*)&Bsm[buf][bn0 + i][kp * 2] = v;
                }
            } else {
                const float4 r0 = bRegF[h * 2 + 0];
                const float4 r1 = bRegF[h * 2 + 1];
                const float a0[4] = {r0.x, r0.y, r0.z, r0.w};
                const float a1[4] = {r1.x, r1.y, r1.z, r1.w};
#pragma unroll
                for (int i = 0; i < 4; ++i) {
                    unsigned v = (unsigned)f2bf(a0[i]) | ((unsigned)f2bf(a1[i]) << 16);
                    *(unsigned*)&Bsm[buf][bn0 + i][kp * 2] = v;
                }
            }
        }
    };

    auto compute = [&](int buf) {
        Frag a[2], b[2];
        const int c0 = hi ? 8 : 0;     // A K-chunks: {c0..c0+7, c0+16..c0+23}
        const int e0 = hi ? 8 : 0;     // B K-chunks: {e0..e0+7, e0+16..e0+23}
#pragma unroll
        for (int t = 0; t < 2; ++t) {
            int m = mw + t * 16 + lr;
            a[t].q[0] = *(const uint4*)&Asm[buf][m][c0];
            a[t].q[1] = *(const uint4*)&Asm[buf][m][c0 + 16];
            int n = nw + t * 16 + lr;
            b[t].q[0] = *(const uint4*)&Bsm[buf][n][e0];
            b[t].q[1] = *(const uint4*)&Bsm[buf][n][e0 + 16];
        }
#pragma unroll
        for (int i = 0; i < 2; ++i)
#pragma unroll
            for (int j = 0; j < 2; ++j)
                acc[i][j] = wmma_bf16(a[i], b[j], acc[i][j]);
    };

    // ---- pipelined main loop: 16 K-steps, 1 barrier each ----
    g2r(0);
    r2s(0);
    __syncthreads();
#pragma unroll 2
    for (int k = 0; k < 16; ++k) {
        if (k < 15) g2r((k + 1) * 32);     // overlap with compute below
        compute(k & 1);
        if (k < 15) {
            r2s((k + 1) & 1);              // prior readers fenced last iter
            __syncthreads();
        }
    }

    // ---- epilogue: bias + store ----
#pragma unroll
    for (int i = 0; i < 2; ++i) {
        int mbase = mblock + mw + i * 16 + hi * 8;
        float4 bz0 = *(const float4*)&bias[mbase];
        float4 bz1 = *(const float4*)&bias[mbase + 4];
        const float bvals[8] = {bz0.x, bz0.y, bz0.z, bz0.w,
                                bz1.x, bz1.y, bz1.z, bz1.w};
#pragma unroll
        for (int j = 0; j < 2; ++j) {
            int nl = nloc + nw + j * 16 + lr;
#pragma unroll
            for (int r = 0; r < 8; ++r) {
                float vv = acc[i][j][r] + bvals[r];
                size_t off = ((size_t)bb * 512 + (mbase + r)) * 4096 + nl;
                if (OUT_BF16)
                    ((unsigned short*)Out)[off] = f2bf(vv);
                else
                    ((float*)Out)[off] = vv;
            }
        }
    }
}

// ---------------------------------------------------------------------------
// Channel attention per (batch, head). One WG (256 thr / 8 waves) per pair.
//   p_h : 32 x 4096 bf16, row stride 4096.
//   att = (p p^T)/64 -> softmax rows -> out = att * p  (bf16 out)
// ---------------------------------------------------------------------------
__global__ __launch_bounds__(256) void attn_kernel(
    const unsigned short* __restrict__ p,
    unsigned short* __restrict__ out)
{
    const int bh   = blockIdx.x;          // 0..255
    const int tid  = threadIdx.x;
    const int lane = tid & 31;
    const int wave = tid >> 5;
    const int lr   = lane & 15;
    const int hi   = lane >> 4;
    const size_t base = (size_t)bh * 32 * 4096;

    __shared__ __align__(16) unsigned short att_bf[32][32]; // softmaxed, bf16
    __shared__ __align__(16) float red[8][1024];            // per-wave partials
    __shared__ __align__(16) float att_s[1024];             // scaled logits
    __shared__ __align__(16) unsigned short pt[512][40];    // p chunk, [n][e]

    // -------- phase A: att = p p^T over n (K split across 8 waves) --------
    v8f acc[2][2] = {};
    const int c0 = hi ? 8 : 0;
    const int e0 = hi ? 8 : 0;
    for (int s = 0; s < 16; ++s) {
        int n0 = wave * 512 + s * 32;
        Frag a[2], b[2];
#pragma unroll
        for (int t = 0; t < 2; ++t) {
            const unsigned short* row = p + base + (size_t)(t * 16 + lr) * 4096 + n0;
            a[t].q[0] = *(const uint4*)(row + c0);        // A chunks
            a[t].q[1] = *(const uint4*)(row + c0 + 16);
            b[t].q[0] = *(const uint4*)(row + e0);        // B chunks (col = row of p)
            b[t].q[1] = *(const uint4*)(row + e0 + 16);
        }
#pragma unroll
        for (int i = 0; i < 2; ++i)
#pragma unroll
            for (int j = 0; j < 2; ++j)
                acc[i][j] = wmma_bf16(a[i], b[j], acc[i][j]);
    }
#pragma unroll
    for (int i = 0; i < 2; ++i)
#pragma unroll
        for (int j = 0; j < 2; ++j)
#pragma unroll
            for (int r = 0; r < 8; ++r) {
                int d = i * 16 + hi * 8 + r;
                int e = j * 16 + lr;
                red[wave][d * 32 + e] = acc[i][j][r];
            }
    __syncthreads();

    // cross-wave reduce + scale (hw^-0.5 = 1/64)
#pragma unroll
    for (int qd = 0; qd < 4; ++qd) {
        int idx = tid + qd * 256;
        float ssum = 0.f;
#pragma unroll
        for (int w = 0; w < 8; ++w) ssum += red[w][idx];
        att_s[idx] = ssum * 0.015625f;
    }
    __syncthreads();

    // -------- phase B: softmax, one row per thread (threads 0..31) --------
    if (tid < 32) {
        float mx = -3.4e38f;
        for (int e = 0; e < 32; ++e) mx = fmaxf(mx, att_s[tid * 32 + e]);
        float sum = 0.f;
        for (int e = 0; e < 32; ++e) sum += __expf(att_s[tid * 32 + e] - mx);
        float inv = 1.f / sum;
        for (int e = 0; e < 32; ++e)
            att_bf[tid][e] = f2bf(__expf(att_s[tid * 32 + e] - mx) * inv);
    }
    __syncthreads();

    // persistent A-fragments of att
    Frag af[2];
#pragma unroll
    for (int t = 0; t < 2; ++t) {
        int d = t * 16 + lr;
        af[t].q[0] = *(const uint4*)&att_bf[d][c0];
        af[t].q[1] = *(const uint4*)&att_bf[d][c0 + 16];
    }

    // -------- phase C: out = att * p, chunked over n --------
    for (int ch = 0; ch < 8; ++ch) {
        int nb = ch * 512;
        __syncthreads();   // protect pt across iterations
        // stage p[0:32, nb:nb+512] transposed into pt[n][e] (coalesced reads)
#pragma unroll
        for (int g = 0; g < 8; ++g) {
            int lin = g * 2048 + tid * 8;
            int e = lin >> 9;
            int n = lin & 511;
            U4S8 v;
            v.q = *(const uint4*)(p + base + (size_t)e * 4096 + nb + n);
#pragma unroll
            for (int i = 0; i < 8; ++i) pt[n + i][e] = v.s[i];
        }
        __syncthreads();
        // 32 n-tiles per chunk, 4 per wave
#pragma unroll
        for (int u = 0; u < 4; ++u) {
            int nt = wave * 4 + u;
            int n = nt * 16 + lr;
            Frag bf;
            bf.q[0] = *(const uint4*)&pt[n][e0];
            bf.q[1] = *(const uint4*)&pt[n][e0 + 16];
            v8f z = {};
            v8f o0 = wmma_bf16(af[0], bf, z);
            v8f o1 = wmma_bf16(af[1], bf, z);
#pragma unroll
            for (int r = 0; r < 8; ++r) {
                int nn = nb + nt * 16 + lr;
                int d0 = hi * 8 + r;
                out[base + (size_t)d0 * 4096 + nn]        = f2bf(o0[r]);
                out[base + (size_t)(16 + d0) * 4096 + nn] = f2bf(o1[r]);
            }
        }
    }
}

// ---------------------------------------------------------------------------
// launcher
// inputs: embedx(16,512,64,64) f32, W_qkv(512,512) f32, b_qkv(512) f32,
//         W_proj(512,512) f32, b_proj(512) f32 -> out(16,512,64,64) f32
// ws: [0,64MB) p_bf16 | [64MB,128MB) attn_out bf16 | [128MB,+1MB) weights bf16
// ---------------------------------------------------------------------------
extern "C" void kernel_launch(void* const* d_in, const int* in_sizes, int n_in,
                              void* d_out, int out_size, void* d_ws, size_t ws_size,
                              hipStream_t stream) {
    const float* embedx = (const float*)d_in[0];
    const float* W_qkv  = (const float*)d_in[1];
    const float* b_qkv  = (const float*)d_in[2];
    const float* W_proj = (const float*)d_in[3];
    const float* b_proj = (const float*)d_in[4];
    float* out = (float*)d_out;

    char* ws = (char*)d_ws;
    unsigned short* p_bf  = (unsigned short*)ws;
    unsigned short* ao_bf = (unsigned short*)(ws + (size_t)67108864);
    unsigned short* wq_bf = (unsigned short*)(ws + (size_t)134217728);
    unsigned short* wp_bf = (unsigned short*)(ws + (size_t)134217728 + 524288);

    cvt_bf16_kernel<<<1024, 256, 0, stream>>>(W_qkv, wq_bf, 262144);
    cvt_bf16_kernel<<<1024, 256, 0, stream>>>(W_proj, wp_bf, 262144);

    dim3 grid(512, 8);   // 512 column blocks x 8 row blocks
    gemm_wmma_kernel<false, true><<<grid, 256, 0, stream>>>(
        wq_bf, (const void*)embedx, b_qkv, (void*)p_bf);

    attn_kernel<<<256, 256, 0, stream>>>(p_bf, ao_bf);

    gemm_wmma_kernel<true, false><<<grid, 256, 0, stream>>>(
        wp_bf, (const void*)ao_bf, b_proj, (void*)out);
}
